// CEAlignmentInformation_40570261078041
// MI455X (gfx1250) — compile-verified, matrix operand order checked by reading, not run on hardware
//
#include <hip/hip_runtime.h>

// ---------------------------------------------------------------------------
// CEAlignmentInformation for MI455X (gfx1250, wave32, WMMA).
// B=1024, X1D=X2D=512, H=512, E=64, C=16, SINK_ITERS=10.
// P (64MB) stays L2-resident (192MB L2); Sinkhorn runs in-place in d_out with
// fused scale-apply + next-sum passes (4 element-passes/iter instead of 6).
// GEMMs: bf16 v_wmma_f32_16x16x32_bf16 (MLP layers), fp32 v_wmma_f32_16x16x4_f32
// for the q1*q2^T affinity (feeds exp -> keep fp32 accuracy).
// ---------------------------------------------------------------------------

#define EPS 1e-8f
#define Bsz 1024
#define Cn  16
#define En  64

typedef __bf16 v16bf __attribute__((ext_vector_type(16)));
typedef float  v8f   __attribute__((ext_vector_type(8)));
typedef float  v4f   __attribute__((ext_vector_type(4)));
typedef float  v2f   __attribute__((ext_vector_type(2)));

// ---- workspace layout (bytes), all offsets 256B aligned -------------------
#define OFF_HA   (0u)                       // [1024,512]  f32  2MB
#define OFF_HB   (2u << 20)                 // [1024,512]  f32  2MB
#define OFF_Q1   (4u << 20)                 // [1024,1024] f32  4MB
#define OFF_Q2   (8u << 20)                 // [1024,1024] f32  4MB
#define OFF_LOG  (12u << 20)                // [1024,16]   64KB
#define OFF_P1   (OFF_LOG  + (64u << 10))
#define OFF_P2   (OFF_P1   + (64u << 10))
#define OFF_P12  (OFF_P2   + (64u << 10))
#define OFF_AALL (OFF_P12  + (64u << 10))
#define OFF_BALL (OFF_AALL + (64u << 10))
#define OFF_RS   (OFF_BALL + (64u << 10))
#define OFF_CS   (OFF_RS   + (64u << 10))
#define OFF_MI1  (OFF_CS   + (64u << 10))
#define OFF_MI2  (OFF_MI1  + 4096u)
#define OFF_MID  (OFF_MI2  + 4096u)
#define OFF_MI12 (OFF_MID  + 4096u)
#define OFF_CS1  (OFF_MI12 + 4096u)
#define OFF_CS2  (OFF_CS1  + 256u)
#define OFF_MAXK (OFF_CS2  + 256u)
// total < 13 MB of d_ws

// ---- helpers --------------------------------------------------------------
__device__ __forceinline__ float waveReduceSum(float v) {
#pragma unroll
  for (int m = 16; m >= 1; m >>= 1) v += __shfl_xor(v, m, 32);
  return v;
}
__device__ __forceinline__ float waveReduceMax(float v) {
#pragma unroll
  for (int m = 16; m >= 1; m >>= 1) v = fmaxf(v, __shfl_xor(v, m, 32));
  return v;
}
__device__ __forceinline__ unsigned fkey(float f) {   // monotonic float->uint
  unsigned u = __float_as_uint(f);
  return (u & 0x80000000u) ? ~u : (u | 0x80000000u);
}
__device__ __forceinline__ float funkey(unsigned k) {
  return __uint_as_float((k & 0x80000000u) ? (k & 0x7fffffffu) : ~k);
}
// bf16 16x16x32 A/B fragment k-mapping (per ISA 7.12.2, 16-bit A 16x32)
__device__ __forceinline__ int kmap16(int lane, int e) {
  return ((e >> 3) << 4) + ((lane >> 4) << 3) + (e & 7);
}

// ---- GEMM: Y = act(X @ W + b), bf16 WMMA, fp32 accum ----------------------
// X is [M,K] (optionally split X1[:,0:K1] | X2[:,K1:K] for the concat case,
// K1 multiple of 32 so a 32-wide k-slab never straddles). A-fragment loads
// are 4x b128; W stays strided (L0/L2 resident, reused by 64 m-tiles).
__global__ void k_gemm_bf16(const float* __restrict__ X1,
                            const float* __restrict__ X2, int K1,
                            const float* __restrict__ W,
                            const float* __restrict__ bias,
                            float* __restrict__ Y,
                            int M, int N, int K, int relu) {
  int wave = threadIdx.x >> 5, lane = threadIdx.x & 31;
  int tiles_n = N >> 4;
  int tile = blockIdx.x * (blockDim.x >> 5) + wave;
  if (tile >= (M >> 4) * tiles_n) return;
  int m0 = (tile / tiles_n) << 4;
  int n0 = (tile % tiles_n) << 4;
  int mA = m0 + (lane & 15);
  int nB = n0 + (lane & 15);
  int hb = (lane >> 4) << 3;                 // half-wave k base
  int K2 = K - K1;
  v8f acc;
  float bv = bias[nB];
#pragma unroll
  for (int r = 0; r < 8; ++r) acc[r] = bv;
  for (int kk = 0; kk < K; kk += 32) {
    const float* Xp = (kk < K1) ? (X1 + (size_t)mA * K1 + kk)
                                : (X2 + (size_t)mA * K2 + (kk - K1));
    v4f x0 = *(const v4f*)(Xp + hb);
    v4f x1 = *(const v4f*)(Xp + hb + 4);
    v4f x2 = *(const v4f*)(Xp + hb + 16);
    v4f x3 = *(const v4f*)(Xp + hb + 20);
    v16bf a, b;
#pragma unroll
    for (int e = 0; e < 4; ++e) {
      a[e]      = (__bf16)x0[e];
      a[e + 4]  = (__bf16)x1[e];
      a[e + 8]  = (__bf16)x2[e];
      a[e + 12] = (__bf16)x3[e];
    }
#pragma unroll
    for (int e = 0; e < 16; ++e)
      b[e] = (__bf16)W[(size_t)(kk + kmap16(lane, e)) * N + nB];
    acc = __builtin_amdgcn_wmma_f32_16x16x32_bf16(false, a, false, b,
                                                  (short)0, acc, false, false);
  }
#pragma unroll
  for (int r = 0; r < 8; ++r) {
    float v = acc[r];
    if (relu) v = fmaxf(v, 0.0f);
    Y[(size_t)(m0 + r + ((lane >> 4) << 3)) * N + nB] = v;
  }
}

// ---- aff[b,d,c] = dot(q1n[b,c,:], q2n[d,c,:]) / 8, fp32 WMMA 16x16x4 ------
// Writes into P region of d_out ([b,d,c], c fastest, scalar stores) and
// folds the per-c global max (wave shuffle + ordered-uint atomicMax).
__global__ void k_aff_f32(const float* __restrict__ q1,
                          const float* __restrict__ q2,
                          float* __restrict__ P,
                          unsigned* __restrict__ mk) {
  int wave = threadIdx.x >> 5, lane = threadIdx.x & 31;
  int tile = blockIdx.x * (blockDim.x >> 5) + wave;   // 16 * 64 * 64 tiles
  int c  = tile >> 12;
  int t  = tile & 4095;
  int m0 = (t >> 6) << 4;
  int n0 = (t & 63) << 4;
  const float* arow = q1 + (size_t)(m0 + (lane & 15)) * 1024 + c * En;
  const float* brow = q2 + (size_t)(n0 + (lane & 15)) * 1024 + c * En;
  int kb = (lane >> 4) << 1;                          // K base per half-wave
  v8f acc = {};
#pragma unroll
  for (int kk = 0; kk < En; kk += 4) {
    v2f a = *(const v2f*)(arow + kk + kb);
    v2f b = *(const v2f*)(brow + kk + kb);
    acc = __builtin_amdgcn_wmma_f32_16x16x4_f32(false, a, false, b,
                                                (short)0, acc, false, false);
  }
  float mloc = -1e30f;
#pragma unroll
  for (int r = 0; r < 8; ++r) {
    int bI = m0 + r + ((lane >> 4) << 3);
    int dI = n0 + (lane & 15);
    float v = acc[r] * 0.125f;                        // 1/sqrt(64)
    P[((size_t)bI * Bsz + dI) * Cn + c] = v;
    mloc = fmaxf(mloc, v);
  }
  mloc = waveReduceMax(mloc);                         // whole wave is one c
  if (lane == 0) atomicMax(&mk[c], fkey(mloc));
}

// ---- small dense layer to 16 logits ---------------------------------------
__global__ void k_logits16(const float* __restrict__ H,
                           const float* __restrict__ W,
                           const float* __restrict__ bias,
                           float* __restrict__ L, int K) {
  int idx = blockIdx.x * blockDim.x + threadIdx.x;   // b*16+c
  if (idx >= Bsz * Cn) return;
  int b = idx >> 4, c = idx & 15;
  const float* hr = H + (size_t)b * K;
  float s = bias[c];
  for (int k = 0; k < K; ++k) s += hr[k] * W[k * Cn + c];
  L[idx] = s;
}

// ---- softmax over C=16 + KL vs prior --------------------------------------
__global__ void k_softmax_mi(const float* __restrict__ L,
                             const float* __restrict__ py,
                             float* __restrict__ p, float* __restrict__ mi) {
  int b = blockIdx.x * blockDim.x + threadIdx.x;
  if (b >= Bsz) return;
  float l[Cn]; float mx = -1e30f;
#pragma unroll
  for (int c = 0; c < Cn; ++c) { l[c] = L[b * Cn + c]; mx = fmaxf(mx, l[c]); }
  float s = 0.f;
#pragma unroll
  for (int c = 0; c < Cn; ++c) { l[c] = __expf(l[c] - mx); s += l[c]; }
  float inv = 1.f / s, m = 0.f;
#pragma unroll
  for (int c = 0; c < Cn; ++c) {
    float pv = l[c] * inv;
    p[b * Cn + c] = pv;
    m += pv * __logf(pv / py[c] + EPS);
  }
  mi[b] = m;
}

// ---- column sums of p [B,16] -> cs[16] ------------------------------------
__global__ void k_colsum16(const float* __restrict__ p, float* __restrict__ cs) {
  __shared__ float ls[256];
  int t = threadIdx.x, c = t & 15, r = t >> 4;
  float s = 0.f;
  for (int b = r; b < Bsz; b += 16) s += p[b * Cn + c];
  ls[t] = s;
  __syncthreads();
  if (t < 16) {
    float acc = 0.f;
    for (int i = 0; i < 16; ++i) acc += ls[c + i * 16];
    cs[c] = acc;
  }
}

__global__ void k_aball(const float* __restrict__ p, const float* __restrict__ cs,
                        float* __restrict__ out) {
  int i = blockIdx.x * blockDim.x + threadIdx.x;
  if (i < Bsz * Cn) out[i] = p[i] / (cs[i & 15] + EPS);
}

// ---- per-(b,c) standardization of q over E=64, ddof=1, in place -----------
__global__ void k_norm(float* __restrict__ q) {
  int wave = threadIdx.x >> 5, lane = threadIdx.x & 31;
  int row = blockIdx.x * (blockDim.x >> 5) + wave;    // b*16+c, 16384 rows
  if (row >= Bsz * Cn) return;
  float* r = q + (size_t)row * En;
  float v0 = r[lane], v1 = r[lane + 32];
  float s  = waveReduceSum(v0 + v1);
  float sq = waveReduceSum(v0 * v0 + v1 * v1);
  float mean = s * (1.f / 64.f);
  float var  = (sq - 64.f * mean * mean) * (1.f / 63.f);
  float inv  = rsqrtf(var + EPS);
  r[lane]      = (v0 - mean) * inv;
  r[lane + 32] = (v1 - mean) * inv;
}

__global__ void k_initmax(unsigned* __restrict__ mk) {
  if (threadIdx.x < Cn) mk[threadIdx.x] = 0u;         // below key(-inf)
}

// ---- fused: P = exp(P - max_c) in place AND initial row-scale -------------
// thread = (b,c); row walk stride 64B; emits rs[b,c] = a_all/(rowsum+eps).
__global__ void k_exp_rowscale(float* __restrict__ P,
                               const unsigned* __restrict__ mk,
                               const float* __restrict__ a_all,
                               float* __restrict__ rs) {
  int idx = blockIdx.x * blockDim.x + threadIdx.x;    // b*16+c
  if (idx >= Bsz * Cn) return;
  int b = idx >> 4, c = idx & 15;
  float mx = funkey(mk[c]);
  float* base = P + (size_t)b * (Bsz * Cn) + c;
  float s = 0.f;
#pragma unroll 8
  for (int d = 0; d < Bsz; ++d) {
    float v = __expf(base[(size_t)d * Cn] - mx);
    base[(size_t)d * Cn] = v;
    s += v;
  }
  rs[idx] = a_all[idx] / (s + EPS);
}

// ---- Sinkhorn pass A: apply row-scale, emit col-scale ---------------------
// thread = (d,c); column walk stride 64KB (L2-resident, prefetched).
__global__ void k_passA(float* __restrict__ P, const float* __restrict__ rs,
                        const float* __restrict__ b_all, float* __restrict__ cs) {
  int idx = blockIdx.x * blockDim.x + threadIdx.x;    // d*16+c
  if (idx >= Bsz * Cn) return;
  int c = idx & 15;
  float* base = P + (size_t)idx;                      // == d*16+c
  float s = 0.f;
#pragma unroll 8
  for (int b = 0; b < Bsz; ++b) {
    if (b + 8 < Bsz)
      __builtin_prefetch(base + (size_t)(b + 8) * (Bsz * Cn), 1, 1);
    float v = base[(size_t)b * (Bsz * Cn)] * rs[b * Cn + c];
    base[(size_t)b * (Bsz * Cn)] = v;
    s += v;
  }
  cs[idx] = b_all[idx] / (s + EPS);
}

// ---- Sinkhorn pass B: apply col-scale, emit next row-scale ----------------
// thread = (b,c). On the last iteration also fuses the mi_x1x2_y reduction:
// the 16 c-lanes of a half-wave hold P[b,d,0..15] -> 4 xor-shuffles give the
// per-(b,d) c-sum for P_cond without an extra 64MB pass.
__global__ void k_passB(float* __restrict__ P, const float* __restrict__ cs,
                        const float* __restrict__ a_all, float* __restrict__ rs,
                        const float* __restrict__ py, float* __restrict__ mi,
                        int last) {
  int idx = blockIdx.x * blockDim.x + threadIdx.x;    // b*16+c
  if (idx >= Bsz * Cn) return;
  int b = idx >> 4, c = idx & 15;
  float* base = P + (size_t)b * (Bsz * Cn) + c;
  float s = 0.f;
  if (!last) {
#pragma unroll 8
    for (int d = 0; d < Bsz; ++d) {
      float v = base[(size_t)d * Cn] * cs[d * Cn + c];
      base[(size_t)d * Cn] = v;
      s += v;
    }
  } else {
    float lpy = __logf(py[c] + EPS);
    float acc = 0.f;
    for (int d = 0; d < Bsz; ++d) {
      float v = base[(size_t)d * Cn] * cs[d * Cn + c];
      base[(size_t)d * Cn] = v;
      s += v;
      float sc = v;                                   // sum over 16 c-lanes
      sc += __shfl_xor(sc, 1, 32); sc += __shfl_xor(sc, 2, 32);
      sc += __shfl_xor(sc, 4, 32); sc += __shfl_xor(sc, 8, 32);
      float pc = v / (sc + EPS);
      acc += v * (__logf(pc + EPS) - lpy);
    }
    acc += __shfl_xor(acc, 1, 32); acc += __shfl_xor(acc, 2, 32);
    acc += __shfl_xor(acc, 4, 32); acc += __shfl_xor(acc, 8, 32);
    if (c == 0) mi[b] = acc;
  }
  rs[idx] = a_all[idx] / (s + EPS);
}

// ---- final scalars: loss + pid_vals ---------------------------------------
__global__ void k_finalize(const float* __restrict__ mi1,
                           const float* __restrict__ mi2,
                           const float* __restrict__ mid,
                           const float* __restrict__ mi12,
                           float* __restrict__ out) {
  __shared__ float ls[32][5];
  int t = threadIdx.x;                                // 1024 threads = B
  float m1 = mi1[t], m2 = mi2[t], m12 = mi12[t], md = mid[t];
  float redv = fmaxf(m1 + m2 - m12, 0.f);
  float u1   = fmaxf(m1 - redv, 0.f);
  float u2   = fmaxf(m2 - redv, 0.f);
  float v[5] = {m12, redv, u1, u2, md};
#pragma unroll
  for (int j = 0; j < 5; ++j) v[j] = waveReduceSum(v[j]);
  if ((t & 31) == 0)
    for (int j = 0; j < 5; ++j) ls[t >> 5][j] = v[j];
  __syncthreads();
  if (t == 0) {
    float s[5] = {0, 0, 0, 0, 0};
    for (int i = 0; i < 32; ++i)
      for (int j = 0; j < 5; ++j) s[j] += ls[i][j];
    float loss = s[0] / (float)Bsz;
    out[0] = loss;
    out[1] = s[1] / (float)Bsz;
    out[2] = s[2] / (float)Bsz;
    out[3] = s[3] / (float)Bsz;
    out[4] = fmaxf(s[4] / (float)Bsz - loss, 0.f);
  }
}

// ---------------------------------------------------------------------------
extern "C" void kernel_launch(void* const* d_in, const int* in_sizes, int n_in,
                              void* d_out, int out_size, void* d_ws, size_t ws_size,
                              hipStream_t stream) {
  const float* x1  = (const float*)d_in[0];
  const float* x2  = (const float*)d_in[1];
  const float* py  = (const float*)d_in[2];
  const float* a1W[4] = {(const float*)d_in[3], (const float*)d_in[5],
                         (const float*)d_in[7], (const float*)d_in[9]};
  const float* a1b[4] = {(const float*)d_in[4], (const float*)d_in[6],
                         (const float*)d_in[8], (const float*)d_in[10]};
  const float* a2W[4] = {(const float*)d_in[11], (const float*)d_in[13],
                         (const float*)d_in[15], (const float*)d_in[17]};
  const float* a2b[4] = {(const float*)d_in[12], (const float*)d_in[14],
                         (const float*)d_in[16], (const float*)d_in[18]};
  const float* d1W0 = (const float*)d_in[19]; const float* d1b0 = (const float*)d_in[20];
  const float* d1W1 = (const float*)d_in[21]; const float* d1b1 = (const float*)d_in[22];
  const float* d2W0 = (const float*)d_in[23]; const float* d2b0 = (const float*)d_in[24];
  const float* d2W1 = (const float*)d_in[25]; const float* d2b1 = (const float*)d_in[26];
  const float* dcW0 = (const float*)d_in[27]; const float* dcb0 = (const float*)d_in[28];
  const float* dcW1 = (const float*)d_in[29]; const float* dcb1 = (const float*)d_in[30];

  char* ws = (char*)d_ws;
  float* hA    = (float*)(ws + OFF_HA);
  float* hB    = (float*)(ws + OFF_HB);
  float* q1    = (float*)(ws + OFF_Q1);
  float* q2    = (float*)(ws + OFF_Q2);
  float* logit = (float*)(ws + OFF_LOG);
  float* p1    = (float*)(ws + OFF_P1);
  float* p2    = (float*)(ws + OFF_P2);
  float* p12   = (float*)(ws + OFF_P12);
  float* aall  = (float*)(ws + OFF_AALL);
  float* ball  = (float*)(ws + OFF_BALL);
  float* rs    = (float*)(ws + OFF_RS);
  float* cs    = (float*)(ws + OFF_CS);
  float* mi1   = (float*)(ws + OFF_MI1);
  float* mi2   = (float*)(ws + OFF_MI2);
  float* mid   = (float*)(ws + OFF_MID);
  float* mi12  = (float*)(ws + OFF_MI12);
  float* cs1   = (float*)(ws + OFF_CS1);
  float* cs2   = (float*)(ws + OFF_CS2);
  unsigned* mk = (unsigned*)(ws + OFF_MAXK);

  float* P = (float*)d_out + 5;   // [B,B,C], 4B-aligned only -> scalar access

  // --- discriminators -> p_y_x*, mi terms ---
  k_gemm_bf16<<<256, 256, 0, stream>>>(x1, x1, 512, d1W0, d1b0, hA, 1024, 512, 512, 1);
  k_logits16<<<64, 256, 0, stream>>>(hA, d1W1, d1b1, logit, 512);
  k_softmax_mi<<<4, 256, 0, stream>>>(logit, py, p1, mi1);

  k_gemm_bf16<<<256, 256, 0, stream>>>(x2, x2, 512, d2W0, d2b0, hA, 1024, 512, 512, 1);
  k_logits16<<<64, 256, 0, stream>>>(hA, d2W1, d2b1, logit, 512);
  k_softmax_mi<<<4, 256, 0, stream>>>(logit, py, p2, mi2);

  k_gemm_bf16<<<256, 256, 0, stream>>>(x1, x2, 512, dcW0, dcb0, hA, 1024, 512, 1024, 1);
  k_logits16<<<64, 256, 0, stream>>>(hA, dcW1, dcb1, logit, 512);
  k_softmax_mi<<<4, 256, 0, stream>>>(logit, py, p12, mid);

  // --- alignment MLPs -> q1, q2 ---
  k_gemm_bf16<<<256, 256, 0, stream>>>(x1, x1, 512, a1W[0], a1b[0], hA, 1024, 512, 512, 1);
  k_gemm_bf16<<<256, 256, 0, stream>>>(hA, hA, 512, a1W[1], a1b[1], hB, 1024, 512, 512, 1);
  k_gemm_bf16<<<256, 256, 0, stream>>>(hB, hB, 512, a1W[2], a1b[2], hA, 1024, 512, 512, 1);
  k_gemm_bf16<<<512, 256, 0, stream>>>(hA, hA, 512, a1W[3], a1b[3], q1, 1024, 1024, 512, 0);

  k_gemm_bf16<<<256, 256, 0, stream>>>(x2, x2, 512, a2W[0], a2b[0], hA, 1024, 512, 512, 1);
  k_gemm_bf16<<<256, 256, 0, stream>>>(hA, hA, 512, a2W[1], a2b[1], hB, 1024, 512, 512, 1);
  k_gemm_bf16<<<256, 256, 0, stream>>>(hB, hB, 512, a2W[2], a2b[2], hA, 1024, 512, 512, 1);
  k_gemm_bf16<<<512, 256, 0, stream>>>(hA, hA, 512, a2W[3], a2b[3], q2, 1024, 1024, 512, 0);

  k_norm<<<2048, 256, 0, stream>>>(q1);
  k_norm<<<2048, 256, 0, stream>>>(q2);

  // --- Sinkhorn marginals ---
  k_colsum16<<<1, 256, 0, stream>>>(p1, cs1);
  k_colsum16<<<1, 256, 0, stream>>>(p2, cs2);
  k_aball<<<64, 256, 0, stream>>>(p1, cs1, aall);
  k_aball<<<64, 256, 0, stream>>>(p2, cs2, ball);

  // --- affinity (fp32 WMMA + fused per-c max), then exp + initial rowscale --
  k_initmax<<<1, 32, 0, stream>>>(mk);
  k_aff_f32<<<8192, 256, 0, stream>>>(q1, q2, P, mk);
  k_exp_rowscale<<<64, 256, 0, stream>>>(P, mk, aall, rs);

  // --- 10 Sinkhorn iterations: 2 fused RW passes each, L2-resident ---
  for (int it = 0; it < 10; ++it) {
    k_passA<<<64, 256, 0, stream>>>(P, rs, ball, cs);
    k_passB<<<64, 256, 0, stream>>>(P, cs, aall, rs, py, mi12, it == 9);
  }

  // --- final scalars (P stays in d_out as the third output) ---
  k_finalize<<<1, 1024, 0, stream>>>(mi1, mi2, mid, mi12, (float*)d_out);
}
